// HyenaOperator_1580547965049
// MI455X (gfx1250) — matrix operand order repaired and checked by reading, not compiled, not run on hardware
//
#include <hip/hip_runtime.h>
#include <hip/hip_bf16.h>

// ---------------------------------------------------------------------------
// Hyena operator for MI455X (gfx1250), wave32 + WMMA.
//
// Pipeline:
//  K1 filter_mlp    : filters[4096][768]  (sin-MLP filter gen + decay)   VALU
//  K2 filter_fft    : Hf[768][8192]       (FFT(filt)/8192)               LDS FFT
//  K3 gemm_wmma<0>  : up = u @ proj_W + proj_b   (16384x2304, K=768)     WMMA
//  K4 shortconv     : w = x0*v, x1        (k=3 causal conv + split)      VALU
//  K5 fftconv       : w <- IFFT(FFT(w) * Hf) + w*filt_bias (per b,c)     LDS FFT
//  K6 gemm_wmma<1>  : y = (w * x1) @ out_W + out_b  (16384x768, K=768)   WMMA
// ---------------------------------------------------------------------------

#define MAXLEN 4096
#define DMODEL 768
#define BATCH  4
#define NPROJ  2304             // (ORDER+1)*D_MODEL
#define MROWS  (BATCH*MAXLEN)   // 16384
#define FFTN   8192

typedef __attribute__((ext_vector_type(16))) _Float16 v16h;
typedef __attribute__((ext_vector_type(8)))  float    v8f;
typedef __attribute__((ext_vector_type(2)))  _Float16 h2;
typedef __attribute__((ext_vector_type(4)))  _Float16 h4;

// ---------------------------------------------------------------------------
// K1: filter generation MLP. One block per time step t, 64 threads.
// ---------------------------------------------------------------------------
__global__ __launch_bounds__(64) void filter_mlp_kernel(
    const float* __restrict__ z, const float* __restrict__ finW,
    const float* __restrict__ finb, const float* __restrict__ freq,
    const float* __restrict__ midW, const float* __restrict__ midb,
    const float* __restrict__ foutW, float* __restrict__ filters)
{
    __shared__ float zr[65];
    __shared__ float h0[64];
    __shared__ float h1[64];
    const int t = blockIdx.x;
    const int j = threadIdx.x;

    for (int i = j; i < 65; i += 64) zr[i] = z[t * 65 + i];
    __syncthreads();

    float a = finb[j];
    for (int i = 0; i < 65; ++i) a += zr[i] * finW[i * 64 + j];
    h0[j] = sinf(freq[j] * a);
    __syncthreads();

    a = midb[j];
    for (int i = 0; i < 64; ++i) a += h0[i] * midW[i * 64 + j];
    h1[j] = sinf(freq[j] * a);
    __syncthreads();

    a = midb[64 + j];
    for (int i = 0; i < 64; ++i) a += h1[i] * midW[64 * 64 + i * 64 + j];
    h0[j] = sinf(freq[j] * a);
    __syncthreads();

    // decay: deltas = linspace(ln(.01)/1.5, ln(.01)/0.3, 768); decay=exp(-t*|d|)
    const float tnorm = (float)t / 4095.0f;
    const float mind = -3.0701134f;   // ln(0.01)/1.5
    const float maxd = -15.3505672f;  // ln(0.01)/0.3
    for (int c = j; c < DMODEL; c += 64) {
        float acc = 0.0f;
        for (int i = 0; i < 64; ++i) acc += h0[i] * foutW[i * DMODEL + c];
        float delta = mind + (maxd - mind) * ((float)c / 767.0f);
        float decay = expf(-tnorm * fabsf(delta));
        filters[(size_t)t * DMODEL + c] = acc * decay;
    }
}

// ---------------------------------------------------------------------------
// In-LDS radix-2 Cooley-Tukey FFT, 8192 points, 256 threads.
// Forward DFT (e^{-2*pi*i*k*t/N}); unscaled.
// ---------------------------------------------------------------------------
__device__ __forceinline__ void fft8192(float* re, float* im, int tid)
{
    // bit-reverse permutation (13 bits)
    for (int i = tid; i < FFTN; i += 256) {
        int j = (int)(__brev((unsigned)i) >> 19);
        if (j > i) {
            float tr = re[i]; re[i] = re[j]; re[j] = tr;
            float ti = im[i]; im[i] = im[j]; im[j] = ti;
        }
    }
    __syncthreads();

    for (int s = 1; s <= 13; ++s) {
        const int m = 1 << s;
        const int half = m >> 1;
        const float ang = -6.28318530717958647692f / (float)m;
        for (int idx = tid; idx < (FFTN / 2); idx += 256) {
            const int g = idx >> (s - 1);
            const int p = idx & (half - 1);
            const int i0 = g * m + p;
            const int i1 = i0 + half;
            float sn, cs;
            __sincosf(ang * (float)p, &sn, &cs);
            const float tr = cs * re[i1] - sn * im[i1];
            const float ti = cs * im[i1] + sn * re[i1];
            re[i1] = re[i0] - tr;
            im[i1] = im[i0] - ti;
            re[i0] += tr;
            im[i0] += ti;
        }
        __syncthreads();
    }
}

// ---------------------------------------------------------------------------
// K2: Hf[c][k] = FFT_8192(filters[:,c], zero-pad) / 8192. One block per channel.
// ---------------------------------------------------------------------------
__global__ __launch_bounds__(256) void filter_fft_kernel(
    const float* __restrict__ filters, float2* __restrict__ Hf)
{
    __shared__ float re[FFTN];
    __shared__ float im[FFTN];
    const int c = blockIdx.x;
    const int tid = threadIdx.x;

    for (int t = tid; t < FFTN; t += 256) {
        re[t] = (t < MAXLEN) ? filters[(size_t)t * DMODEL + c] : 0.0f;
        im[t] = 0.0f;
    }
    __syncthreads();
    fft8192(re, im, tid);
    const float scale = 1.0f / (float)FFTN;
    float2* H = Hf + (size_t)c * FFTN;
    for (int k = tid; k < FFTN; k += 256)
        H[k] = make_float2(re[k] * scale, im[k] * scale);
}

// ---------------------------------------------------------------------------
// K3/K6: tiled f16-WMMA GEMM, C[M][N] = A[M][K] @ B[K][N] + bias[N].
// MODE==1: A element = A0*A1 (fused elementwise product).
// Block tile 128x128x64, 256 threads (8 waves), wave tile 64x32 (4x2 WMMAs).
// LDS: As[m][k] (16KB) and Bs[n][k] *transposed* (16KB) so both A and B
// fragments are contiguous 16B/32B runs -> ds_load_b128.
// ---------------------------------------------------------------------------
template <int MODE>
__global__ __launch_bounds__(256) void gemm_wmma_kernel(
    const float* __restrict__ A0, const float* __restrict__ A1,
    const float* __restrict__ B, const float* __restrict__ bias,
    float* __restrict__ C, int M, int N, int K)
{
    __shared__ _Float16 As[128][64];  // [m][k]
    __shared__ _Float16 Bs[128][64];  // [n][k]  (transposed tile)

    const int tid  = threadIdx.x;
    const int lane = tid & 31;
    const int wave = tid >> 5;
    const int wm   = (wave & 1) * 64;   // wave row offset in block tile
    const int wn   = (wave >> 1) * 32;  // wave col offset in block tile
    const int m0   = blockIdx.y * 128;
    const int n0   = blockIdx.x * 128;

    const int hi  = lane >> 4;   // half-wave select
    const int lml = lane & 15;

    v8f acc[4][2];
#pragma unroll
    for (int mt = 0; mt < 4; ++mt)
#pragma unroll
        for (int nt = 0; nt < 2; ++nt)
            acc[mt][nt] = (v8f){0.f, 0.f, 0.f, 0.f, 0.f, 0.f, 0.f, 0.f};

    for (int k0 = 0; k0 < K; k0 += 64) {
        // ---- stage A tile: 128x64 f32 -> f16, float4 loads, h4 LDS stores
#pragma unroll
        for (int it = 0; it < 8; ++it) {
            const int q  = tid + 256 * it;
            const int r  = q >> 4;
            const int c4 = (q & 15) << 2;
            const size_t gi = (size_t)(m0 + r) * K + (k0 + c4);
            const float4 v4 = *(const float4*)&A0[gi];
            float x0 = v4.x, x1 = v4.y, x2 = v4.z, x3 = v4.w;
            if (MODE) {
                const float4 w4 = *(const float4*)&A1[gi];
                x0 *= w4.x; x1 *= w4.y; x2 *= w4.z; x3 *= w4.w;
            }
            *(h4*)&As[r][c4] =
                (h4){(_Float16)x0, (_Float16)x1, (_Float16)x2, (_Float16)x3};
        }
        // ---- stage B tile transposed: read 2 rows of K as float4 along N,
        //      pack (k,k+1) pairs, b32 LDS stores into Bs[n][k]
#pragma unroll
        for (int it = 0; it < 4; ++it) {
            const int e  = tid + 256 * it;
            const int kp = e >> 5;            // k pair index 0..31
            const int n4 = (e & 31) << 2;     // n group 0..124
            const int k  = 2 * kp;
            const float4 b0 = *(const float4*)&B[(size_t)(k0 + k) * N + (n0 + n4)];
            const float4 b1 = *(const float4*)&B[(size_t)(k0 + k + 1) * N + (n0 + n4)];
            *(h2*)&Bs[n4 + 0][k] = (h2){(_Float16)b0.x, (_Float16)b1.x};
            *(h2*)&Bs[n4 + 1][k] = (h2){(_Float16)b0.y, (_Float16)b1.y};
            *(h2*)&Bs[n4 + 2][k] = (h2){(_Float16)b0.z, (_Float16)b1.z};
            *(h2*)&Bs[n4 + 3][k] = (h2){(_Float16)b0.w, (_Float16)b1.w};
        }
        // ---- prefetch next K tile into cache (global_prefetch_b8)
        if (k0 + 64 < K) {
            __builtin_prefetch(&A0[(size_t)(m0 + (tid >> 4)) * K + (k0 + 64)], 0, 3);
            __builtin_prefetch(&B[(size_t)(k0 + 64 + (tid >> 7)) * N + n0 + (tid & 127)], 0, 3);
        }
        __syncthreads();

#pragma unroll
        for (int ks = 0; ks < 2; ++ks) {
            const int kb = ks * 32;
            // B fragments: lane = col n; element j holds K = kb + 16*hi + j
            v16h bfrag[2];
#pragma unroll
            for (int nt = 0; nt < 2; ++nt) {
                const int n = wn + nt * 16 + lml;
#pragma unroll
                for (int j = 0; j < 16; ++j)
                    bfrag[nt][j] = Bs[n][kb + 16 * hi + j];
            }
#pragma unroll
            for (int mt = 0; mt < 4; ++mt) {
                // A fragment: element j holds K = kb + 16*hi + j
                v16h afrag;
                const int m = wm + mt * 16 + lml;
#pragma unroll
                for (int j = 0; j < 16; ++j)
                    afrag[j] = As[m][kb + 16 * hi + j];
#pragma unroll
                for (int nt = 0; nt < 2; ++nt) {
                    acc[mt][nt] = __builtin_amdgcn_wmma_f32_16x16x32_f16(
                        false, afrag, false, bfrag[nt], (short)0, acc[mt][nt],
                        false, false);
                }
            }
        }
        __syncthreads();
    }

    // store with fused bias; C/D layout: vgpr r -> M = r + 8*hi, lane -> N
#pragma unroll
    for (int mt = 0; mt < 4; ++mt) {
#pragma unroll
        for (int nt = 0; nt < 2; ++nt) {
            const int n = n0 + wn + nt * 16 + lml;
            const float bv = bias[n];
#pragma unroll
            for (int r = 0; r < 8; ++r) {
                const int m = m0 + wm + mt * 16 + r + 8 * hi;
                C[(size_t)m * N + n] = acc[mt][nt][r] + bv;
            }
        }
    }
}

// ---------------------------------------------------------------------------
// K4: width-3 causal conv on `up`, split into (x0,x1,v); emit w=x0*v and x1.
// Vectorized: 4 channels per thread (float4).
// ---------------------------------------------------------------------------
__global__ __launch_bounds__(256) void shortconv_kernel(
    const float* __restrict__ up, const float* __restrict__ convW,
    const float* __restrict__ convb, float* __restrict__ w,
    float* __restrict__ x1)
{
    const size_t idx4 = (size_t)blockIdx.x * blockDim.x + threadIdx.x;
    const size_t total4 = (size_t)BATCH * MAXLEN * (DMODEL / 4);
    if (idx4 >= total4) return;
    const int c  = (int)(idx4 % (DMODEL / 4)) * 4;
    const int t  = (int)((idx4 / (DMODEL / 4)) % MAXLEN);
    const int b  = (int)(idx4 / ((size_t)(DMODEL / 4) * MAXLEN));
    const float* upb = up + (size_t)b * MAXLEN * NPROJ;

    float4 uc[3];
#pragma unroll
    for (int band = 0; band < 3; ++band) {
        const int cb = band * DMODEL + c;
        const float4 cbv = *(const float4*)&convb[cb];
        float4 acc = cbv;
#pragma unroll
        for (int jj = 0; jj < 3; ++jj) {
            const int tt = t - 2 + jj;
            if (tt >= 0) {
                const float4 wv = *(const float4*)&convW[jj * NPROJ + cb];
                const float4 uv = *(const float4*)&upb[(size_t)tt * NPROJ + cb];
                acc.x += wv.x * uv.x; acc.y += wv.y * uv.y;
                acc.z += wv.z * uv.z; acc.w += wv.w * uv.w;
            }
        }
        uc[band] = acc;
    }
    float4 wout, x1out = uc[1];
    wout.x = uc[0].x * uc[2].x; wout.y = uc[0].y * uc[2].y;
    wout.z = uc[0].z * uc[2].z; wout.w = uc[0].w * uc[2].w;
    *(float4*)&w[idx4 * 4]  = wout;   // x0 * v
    *(float4*)&x1[idx4 * 4] = x1out;
}

// ---------------------------------------------------------------------------
// K5: per-(b,c) circular conv via FFT, written back into w in place.
// y[t] = Re(IFFT(FFT(w_pad) * Hf[c]))[t] + w[t]*filt_bias[c]
// IFFT via conjugate trick: ifft(X) = conj(fft(conj(X))) (Hf carries 1/N).
// ---------------------------------------------------------------------------
__global__ __launch_bounds__(256) void fftconv_kernel(
    float* __restrict__ w, const float2* __restrict__ Hf,
    const float* __restrict__ filt_bias)
{
    __shared__ float re[FFTN];
    __shared__ float im[FFTN];
    const int c = blockIdx.x;
    const int b = blockIdx.y;
    const int tid = threadIdx.x;
    float* wc = w + (size_t)b * MAXLEN * DMODEL + c;

    // cache this (b,c) column in registers (16 vals/thread) while staging LDS
    float wval[16];
    {
        int q = 0;
        for (int t = tid; t < MAXLEN; t += 256, ++q) {
            const float v = wc[(size_t)t * DMODEL];
            wval[q] = v;
            re[t] = v;
            im[t] = 0.0f;
        }
        for (int t = MAXLEN + tid; t < FFTN; t += 256) {
            re[t] = 0.0f;
            im[t] = 0.0f;
        }
    }
    __syncthreads();
    fft8192(re, im, tid);

    const float2* H = Hf + (size_t)c * FFTN;
    for (int k = tid; k < FFTN; k += 256) {
        const float2 h = H[k];
        const float xr = re[k], xi = im[k];
        const float pr = xr * h.x - xi * h.y;
        const float pi = xr * h.y + xi * h.x;
        re[k] = pr;       // conj(P) for inverse transform
        im[k] = -pi;
    }
    __syncthreads();
    fft8192(re, im, tid);

    const float bias = filt_bias[c];
    int q = 0;
    for (int t = tid; t < MAXLEN; t += 256, ++q)
        wc[(size_t)t * DMODEL] = re[t] + wval[q] * bias;
}

// ---------------------------------------------------------------------------
// Launcher
// ---------------------------------------------------------------------------
extern "C" void kernel_launch(void* const* d_in, const int* in_sizes, int n_in,
                              void* d_out, int out_size, void* d_ws, size_t ws_size,
                              hipStream_t stream)
{
    const float* u         = (const float*)d_in[0];
    const float* z         = (const float*)d_in[1];
    const float* fin_W     = (const float*)d_in[2];
    const float* fin_b     = (const float*)d_in[3];
    const float* freq      = (const float*)d_in[4];
    const float* mid_W     = (const float*)d_in[5];
    const float* mid_b     = (const float*)d_in[6];
    const float* fout_W    = (const float*)d_in[7];
    const float* proj_W    = (const float*)d_in[8];
    const float* proj_b    = (const float*)d_in[9];
    const float* conv_W    = (const float*)d_in[10];
    const float* conv_b    = (const float*)d_in[11];
    const float* filt_bias = (const float*)d_in[12];
    const float* out_W     = (const float*)d_in[13];
    const float* out_b     = (const float*)d_in[14];
    float* y = (float*)d_out;

    // workspace layout (all chunks naturally 256B aligned)
    char* ws = (char*)d_ws;
    size_t off = 0;
    float*  filters = (float*)(ws + off);  off += (size_t)MAXLEN * DMODEL * 4;   // 12.6 MB
    float2* Hf      = (float2*)(ws + off); off += (size_t)DMODEL * FFTN * 8;     // 50.3 MB
    float*  up      = (float*)(ws + off);  off += (size_t)MROWS * NPROJ * 4;     // 151  MB
    float*  wbuf    = (float*)(ws + off);  off += (size_t)MROWS * DMODEL * 4;    // 50.3 MB
    float*  x1buf   = (float*)(ws + off);  off += (size_t)MROWS * DMODEL * 4;    // 50.3 MB
    if (ws_size < off) return;  // workspace too small; nothing sane to do

    // K1: filter MLP + decay
    filter_mlp_kernel<<<MAXLEN, 64, 0, stream>>>(
        z, fin_W, fin_b, freq, mid_W, mid_b, fout_W, filters);

    // K2: filter spectrum (768 channel FFTs)
    filter_fft_kernel<<<DMODEL, 256, 0, stream>>>(filters, Hf);

    // K3: up = u @ proj_W + proj_b   [16384 x 2304, K=768]
    gemm_wmma_kernel<0><<<dim3(NPROJ / 128, MROWS / 128), 256, 0, stream>>>(
        u, nullptr, proj_W, proj_b, up, MROWS, NPROJ, DMODEL);

    // K4: short conv, split, products (float4 over channels)
    {
        const size_t total4 = (size_t)BATCH * MAXLEN * (DMODEL / 4);
        shortconv_kernel<<<(unsigned)((total4 + 255) / 256), 256, 0, stream>>>(
            up, conv_W, conv_b, wbuf, x1buf);
    }

    // K5: long conv via FFT (3072 seq FFT pairs), in-place into wbuf
    fftconv_kernel<<<dim3(DMODEL, BATCH), 256, 0, stream>>>(wbuf, Hf, filt_bias);

    // K6: y = (wbuf * x1) @ out_W + out_b   [16384 x 768, K=768]
    gemm_wmma_kernel<1><<<dim3(DMODEL / 128, MROWS / 128), 256, 0, stream>>>(
        wbuf, x1buf, out_W, out_b, y, MROWS, DMODEL, DMODEL);
}
